// SelfAttention_23021024706947
// MI455X (gfx1250) — matrix operand order repaired and checked by reading, not compiled, not run on hardware
//
#include <hip/hip_runtime.h>

// ---------------- problem constants ----------------
#define BB    2
#define DIMC  256
#define XX    48
#define NN    2304          // 48*48
#define HEADS 8
#define DH    64
#define INNER 512           // HEADS*DH
#define NKEYS 2305          // null + N
#define NP    2336          // padded keys (73 * 32)
#define MASK_SELF (-100.0f)

typedef __attribute__((ext_vector_type(16))) _Float16 v16h;
typedef __attribute__((ext_vector_type(8)))  float    v8f;

// ---------------- WMMA helpers ----------------
__device__ __forceinline__ v8f wmma_f16(v16h a, v16h b, v8f c) {
  // D = A(16x32 f16) * B(32x16 f16) + C(16x16 f32)
  return __builtin_amdgcn_wmma_f32_16x16x32_f16(
      false, a, false, b, (short)0, c, false, false);
}

// A fragment: 16x32 f16, row-major source with leading dim `ld` (halves).
// lane holds row `row`; half pairs at K = kbase + (lane>=16?16:0) + 2*p  (contiguous)
__device__ __forceinline__ v16h load_a_frag(const _Float16* __restrict__ src,
                                            int ld, int row, int kbase, int lane) {
  const _Float16* r = src + (size_t)row * ld;
  int hi = (lane & 16) ? 16 : 0;
  v16h a;
#pragma unroll
  for (int p = 0; p < 8; ++p) {
    int k = kbase + hi + 2 * p;
    a[2 * p]     = r[k];
    a[2 * p + 1] = r[k + 1];
  }
  return a;
}

// B fragment: 32x16 f16 from "Bt" storage Bt[col][k] (k contiguous, ld halves).
// lane holds column `col`; halves j=0..15 at K = kbase + 16*(lane/16) + j  (contiguous)
__device__ __forceinline__ v16h load_bt_frag(const _Float16* __restrict__ bt,
                                             int ld, int col, int kbase, int lane) {
  const _Float16* p = bt + (size_t)col * ld + kbase + ((lane & 16) ? 16 : 0);
  return *(const v16h*)p;
}

// ---------------- small utility kernels ----------------
__global__ void f32_to_f16_k(const float* __restrict__ src, _Float16* __restrict__ dst, int n) {
  int i = blockIdx.x * blockDim.x + threadIdx.x;
  if (i < n) dst[i] = (_Float16)src[i];
}

__global__ void fill_zero_h_k(_Float16* __restrict__ dst, int n) {
  int i = blockIdx.x * blockDim.x + threadIdx.x;
  if (i < n) dst[i] = (_Float16)0.0f;
}

// ChannelRMSNorm: per (b,n) over 256 channels; writes f16 normed_t[b][n][c]
__global__ void rmsnorm_k(const float* __restrict__ fmap, const float* __restrict__ gamma,
                          _Float16* __restrict__ nt) {
  int idx = blockIdx.x * blockDim.x + threadIdx.x;   // b*NN + n
  if (idx >= BB * NN) return;
  int b = idx / NN, n = idx % NN;
  const float* col = fmap + (size_t)b * DIMC * NN + n;
  float ss = 0.0f;
  for (int c = 0; c < DIMC; ++c) { float v = col[(size_t)c * NN]; ss += v * v; }
  float l2  = sqrtf(ss);
  float inv = 16.0f / fmaxf(l2, 1e-12f);             // sqrt(256)=16
  _Float16* out = nt + (size_t)idx * DIMC;
  for (int c = 0; c < DIMC; ++c) out[c] = (_Float16)(col[(size_t)c * NN] * inv * gamma[c]);
}

// write null kv into key row 0 / vt column 0
__global__ void init_nullkv_k(const float* __restrict__ null_kv,
                              _Float16* __restrict__ kf, _Float16* __restrict__ vtf) {
  int i = blockIdx.x * blockDim.x + threadIdx.x;     // b*8*64
  if (i >= BB * HEADS * DH) return;
  int d = i & 63; int h = (i >> 6) & 7; int b = i >> 9;
  int bh = b * HEADS + h;
  kf [((size_t)bh * NP + 0) * DH + d]  = (_Float16)null_kv[h * DH + d];          // null_kv[0]
  vtf[((size_t)bh * DH + d) * NP + 0]  = (_Float16)null_kv[INNER + h * DH + d];  // null_kv[1]
}

// sum of squares over 64-element f16 rows
__global__ void rowsq_k(const _Float16* __restrict__ src, float* __restrict__ dst, int rows) {
  int i = blockIdx.x * blockDim.x + threadIdx.x;
  if (i >= rows) return;
  const _Float16* p = src + (size_t)i * DH;
  float s = 0.0f;
  for (int d = 0; d < DH; ++d) { float v = (float)p[d]; s += v * v; }
  dst[i] = s;
}

// ---------------- projection GEMM: [w_qk;w_v](1024x256) x normed^T ----------------
// grid: (NN/16, 1024/16, B), block: 32 (one wave)
__global__ __launch_bounds__(32) void proj_gemm_k(
    const _Float16* __restrict__ wh,   // [1024][256]
    const _Float16* __restrict__ nt,   // [B][NN][256]
    _Float16* __restrict__ qf,         // [B*H][NN][64]
    _Float16* __restrict__ kf,         // [B*H][NP][64]
    _Float16* __restrict__ vtf) {      // [B*H][64][NP]
  int lane = threadIdx.x;
  int ntile = blockIdx.x, ot = blockIdx.y, b = blockIdx.z;
  const _Float16* bt = nt + (size_t)b * NN * DIMC;
  int arow = ot * 16 + (lane & 15);
  int bcol = ntile * 16 + (lane & 15);
  v8f acc = {};
#pragma unroll
  for (int kc = 0; kc < DIMC; kc += 32) {
    v16h a  = load_a_frag(wh, DIMC, arow, kc, lane);
    v16h bb = load_bt_frag(bt, DIMC, bcol, kc, lane);
    acc = wmma_f16(a, bb, acc);
  }
  int rbase = (lane & 16) ? 8 : 0;
  int n = ntile * 16 + (lane & 15);
#pragma unroll
  for (int r = 0; r < 8; ++r) {
    int o = ot * 16 + r + rbase;
    _Float16 hv = (_Float16)acc[r];
    if (o < INNER) {                       // qk rows
      int h = o >> 6, d = o & 63;
      int bh = b * HEADS + h;
      qf[((size_t)bh * NN + n) * DH + d]       = hv;
      kf[((size_t)bh * NP + (n + 1)) * DH + d] = hv;
    } else {                               // v rows -> transposed
      int o2 = o - INNER; int h = o2 >> 6, d = o2 & 63;
      int bh = b * HEADS + h;
      vtf[((size_t)bh * DH + d) * NP + (n + 1)] = hv;
    }
  }
}

// ---------------- flash-style attention ----------------
// grid: (NN/16 query tiles, B*H), block: 32 (one wave)
__global__ __launch_bounds__(32) void attn_k(
    const _Float16* __restrict__ qf, const _Float16* __restrict__ kf,
    const _Float16* __restrict__ vtf, const float* __restrict__ q2,
    const float* __restrict__ k2, _Float16* __restrict__ aout) {
  __shared__ _Float16 pls[16 * 32];
  int lane = threadIdx.x;
  int qt = blockIdx.x, bh = blockIdx.y;
  const _Float16* qrow = qf  + (size_t)bh * NN * DH;
  const _Float16* krow = kf  + (size_t)bh * NP * DH;
  const _Float16* vrow = vtf + (size_t)bh * DH * NP;
  const float* q2p = q2 + (size_t)bh * NN + qt * 16;
  const float* k2p = k2 + (size_t)bh * NP;

  int m = lane & 15;
  int rbase = (lane & 16) ? 8 : 0;
  v16h aq0 = load_a_frag(qrow, DH, qt * 16 + m, 0, lane);
  v16h aq1 = load_a_frag(qrow, DH, qt * 16 + m, 32, lane);

  float q2r[8];
#pragma unroll
  for (int r = 0; r < 8; ++r) q2r[r] = q2p[r + rbase];

  float rowmax[8], rowsum[8];
  v8f oacc[4];
#pragma unroll
  for (int r = 0; r < 8; ++r) { rowmax[r] = -3.0e38f; rowsum[r] = 0.0f; }
#pragma unroll
  for (int cd = 0; cd < 4; ++cd) oacc[cd] = (v8f){};

  const float scale = 0.125f;  // 64^-0.5

  for (int kb = 0; kb < NP; kb += 32) {
    float sims[2][8];
#pragma unroll
    for (int t = 0; t < 2; ++t) {
      int kcol = kb + t * 16 + (lane & 15);
      v16h b0 = load_bt_frag(krow, DH, kcol, 0, lane);
      v16h b1 = load_bt_frag(krow, DH, kcol, 32, lane);
      v8f c = {};
      c = wmma_f16(aq0, b0, c);
      c = wmma_f16(aq1, b1, c);
      float k2c = k2p[kcol];
#pragma unroll
      for (int r = 0; r < 8; ++r) {
        float d2 = q2r[r] + k2c - 2.0f * c[r];
        d2 = d2 > 0.0f ? d2 : 0.0f;
        float s = -sqrtf(d2) * scale;
        int mg = qt * 16 + r + rbase;
        if (kcol == mg + 1)  s = MASK_SELF;   // self mask (keys left-padded by null)
        if (kcol >= NKEYS)   s = -1.0e30f;    // padding
        sims[t][r] = s;
      }
    }
    // online softmax: row reductions across the 16 lanes of each half-wave
    float newmax[8], alpha[8];
#pragma unroll
    for (int r = 0; r < 8; ++r) {
      float mx = fmaxf(sims[0][r], sims[1][r]);
#pragma unroll
      for (int off = 1; off < 16; off <<= 1)
        mx = fmaxf(mx, __shfl_xor(mx, off, 32));
      newmax[r] = fmaxf(rowmax[r], mx);
      alpha[r]  = __expf(rowmax[r] - newmax[r]);
      rowmax[r] = newmax[r];
    }
#pragma unroll
    for (int r = 0; r < 8; ++r) {
      float p0 = __expf(sims[0][r] - newmax[r]);
      float p1 = __expf(sims[1][r] - newmax[r]);
      pls[(r + rbase) * 32 + (lane & 15)]      = (_Float16)p0;
      pls[(r + rbase) * 32 + 16 + (lane & 15)] = (_Float16)p1;
      float s = p0 + p1;
#pragma unroll
      for (int off = 1; off < 16; off <<= 1)
        s += __shfl_xor(s, off, 32);
      rowsum[r] = rowsum[r] * alpha[r] + s;
    }
    __syncthreads();
    v16h pa = load_a_frag(pls, 32, lane & 15, 0, lane);
#pragma unroll
    for (int cd = 0; cd < 4; ++cd) {
#pragma unroll
      for (int r = 0; r < 8; ++r) oacc[cd][r] *= alpha[r];
      v16h bv = load_bt_frag(vrow, NP, cd * 16 + (lane & 15), kb, lane);
      oacc[cd] = wmma_f16(pa, bv, oacc[cd]);
    }
    __syncthreads();
  }

  int b = bh >> 3, h = bh & 7;
#pragma unroll
  for (int cd = 0; cd < 4; ++cd) {
#pragma unroll
    for (int r = 0; r < 8; ++r) {
      int n = qt * 16 + r + rbase;
      int inner = h * DH + cd * 16 + (lane & 15);
      aout[((size_t)b * NN + n) * INNER + inner] = (_Float16)(oacc[cd][r] / rowsum[r]);
    }
  }
}

// ---------------- output GEMM: w_out(256x512) x attn_out^T ----------------
// grid: (NN/16, 256/16, B), block: 32
__global__ __launch_bounds__(32) void out_gemm_k(
    const _Float16* __restrict__ wouth,  // [256][512]
    const _Float16* __restrict__ aout,   // [B][NN][512]
    float* __restrict__ out) {           // [B][256][NN]
  int lane = threadIdx.x;
  int ntile = blockIdx.x, ot = blockIdx.y, b = blockIdx.z;
  const _Float16* bt = aout + (size_t)b * NN * INNER;
  int arow = ot * 16 + (lane & 15);
  int bcol = ntile * 16 + (lane & 15);
  v8f acc = {};
#pragma unroll
  for (int kc = 0; kc < INNER; kc += 32) {
    v16h a  = load_a_frag(wouth, INNER, arow, kc, lane);
    v16h bb = load_bt_frag(bt, INNER, bcol, kc, lane);
    acc = wmma_f16(a, bb, acc);
  }
  int rbase = (lane & 16) ? 8 : 0;
  int n = ntile * 16 + (lane & 15);
#pragma unroll
  for (int r = 0; r < 8; ++r) {
    int o = ot * 16 + r + rbase;
    out[((size_t)b * DIMC + o) * NN + n] = acc[r];
  }
}

// ---------------- host side ----------------
extern "C" void kernel_launch(void* const* d_in, const int* in_sizes, int n_in,
                              void* d_out, int out_size, void* d_ws, size_t ws_size,
                              hipStream_t stream) {
  (void)in_sizes; (void)n_in; (void)out_size; (void)ws_size;
  const float* fmap    = (const float*)d_in[0];
  const float* gamma   = (const float*)d_in[1];
  const float* w_qk    = (const float*)d_in[2];
  const float* w_v     = (const float*)d_in[3];
  const float* null_kv = (const float*)d_in[4];
  const float* w_out   = (const float*)d_in[5];
  float* out = (float*)d_out;

  char* ws = (char*)d_ws;
  size_t off = 0;
  auto alloc = [&](size_t bytes) -> void* {
    void* p = ws + off;
    off += (bytes + 255) & ~(size_t)255;
    return p;
  };

  _Float16* w_h      = (_Float16*)alloc((size_t)1024 * DIMC * 2);          // [1024][256]
  _Float16* w_out_h  = (_Float16*)alloc((size_t)DIMC * INNER * 2);         // [256][512]
  _Float16* normed_t = (_Float16*)alloc((size_t)BB * NN * DIMC * 2);       // [B][N][256]
  _Float16* q_f16    = (_Float16*)alloc((size_t)BB * HEADS * NN * DH * 2); // [BH][N][64]
  _Float16* k_f16    = (_Float16*)alloc((size_t)BB * HEADS * NP * DH * 2); // [BH][NP][64]
  _Float16* vt_f16   = (_Float16*)alloc((size_t)BB * HEADS * DH * NP * 2); // [BH][64][NP]
  _Float16* attn_out = (_Float16*)alloc((size_t)BB * NN * INNER * 2);      // [B][N][512]
  float*    q2       = (float*)alloc((size_t)BB * HEADS * NN * 4);
  float*    k2       = (float*)alloc((size_t)BB * HEADS * NP * 4);

  const int T = 256;
  // 1) convert weights to f16 (w_h = [w_qk ; w_v])
  f32_to_f16_k<<<(INNER * DIMC + T - 1) / T, T, 0, stream>>>(w_qk, w_h, INNER * DIMC);
  f32_to_f16_k<<<(INNER * DIMC + T - 1) / T, T, 0, stream>>>(w_v, w_h + (size_t)INNER * DIMC, INNER * DIMC);
  f32_to_f16_k<<<(DIMC * INNER + T - 1) / T, T, 0, stream>>>(w_out, w_out_h, DIMC * INNER);
  // 2) zero padded K / Vt buffers
  {
    int nk = BB * HEADS * NP * DH;
    fill_zero_h_k<<<(nk + T - 1) / T, T, 0, stream>>>(k_f16, nk);
    fill_zero_h_k<<<(nk + T - 1) / T, T, 0, stream>>>(vt_f16, nk);
  }
  // 3) RMSNorm
  rmsnorm_k<<<(BB * NN + T - 1) / T, T, 0, stream>>>(fmap, gamma, normed_t);
  // 4) null kv
  init_nullkv_k<<<(BB * HEADS * DH + T - 1) / T, T, 0, stream>>>(null_kv, k_f16, vt_f16);
  // 5) projections (WMMA)
  proj_gemm_k<<<dim3(NN / 16, 1024 / 16, BB), 32, 0, stream>>>(w_h, normed_t, q_f16, k_f16, vt_f16);
  // 6) squared norms
  rowsq_k<<<(BB * HEADS * NN + T - 1) / T, T, 0, stream>>>(q_f16, q2, BB * HEADS * NN);
  rowsq_k<<<(BB * HEADS * NP + T - 1) / T, T, 0, stream>>>(k_f16, k2, BB * HEADS * NP);
  // 7) attention (WMMA, flash-style)
  attn_k<<<dim3(NN / 16, BB * HEADS), 32, 0, stream>>>(q_f16, k_f16, vt_f16, q2, k2, attn_out);
  // 8) output projection (WMMA)
  out_gemm_k<<<dim3(NN / 16, DIMC / 16, BB), 32, 0, stream>>>(w_out_h, attn_out, out);
}
